// MultiHeadSelfAttention_6914897347145
// MI455X (gfx1250) — matrix operand order repaired and checked by reading, not compiled, not run on hardware
//
#include <hip/hip_runtime.h>
#include <stdint.h>

#define B_ 4
#define T_ 2048
#define D_ 1024
#define H_ 16
#define P_ 64
#define HP_ (H_ * P_)

typedef __attribute__((ext_vector_type(16))) __bf16        v16bf;
typedef __attribute__((ext_vector_type(8)))  float          v8f;
typedef __attribute__((ext_vector_type(8)))  unsigned int   v8u;
typedef __attribute__((ext_vector_type(16))) unsigned short v16us;
typedef __attribute__((ext_vector_type(4)))  unsigned int   v4u;
typedef __attribute__((ext_vector_type(8)))  int            v8i;
typedef __attribute__((ext_vector_type(4)))  int            v4i;

#if defined(__has_builtin)
#if __has_builtin(__builtin_amdgcn_tensor_load_to_lds) && \
    __has_builtin(__builtin_amdgcn_s_wait_tensorcnt)
#define HAVE_TDM 1
#endif
#endif

static __device__ __forceinline__ v8f wmma_bf16(v16bf a, v16bf b, v8f c) {
  // D(16x16 f32) = A(16x32 bf16) * B(32x16 bf16) + C
  return __builtin_amdgcn_wmma_f32_16x16x32_bf16(false, a, false, b, (short)0, c,
                                                 false, false);
}

static __device__ __forceinline__ unsigned short f2bf(float f) {
  return __builtin_bit_cast(unsigned short, (__bf16)f);
}

// ---------------------------------------------------------------------------
// fp32 -> bf16 bit converter (one-time pass over x and the weights)
// ---------------------------------------------------------------------------
__global__ __launch_bounds__(256)
void cvt_f32_bf16(const float* __restrict__ in, unsigned short* __restrict__ out,
                  int n) {
  int i = blockIdx.x * 1024 + threadIdx.x;
#pragma unroll
  for (int j = 0; j < 4; ++j, i += 256)
    if (i < n) out[i] = f2bf(in[i]);
}

#ifdef HAVE_TDM
// Build a 2D TDM descriptor (D#) and issue TENSOR_LOAD_TO_LDS.
// Layout per CDNA5 ISA; data_size = 2 bytes.
// This toolchain's builtin takes 6 args (g0,g1,g2,g3,g4,cpol).
static __device__ __forceinline__ void tdm_load_2d(unsigned lds_addr,
                                                   unsigned long long gaddr,
                                                   unsigned tensor_d0,
                                                   unsigned tensor_d1,
                                                   unsigned tile_d0,
                                                   unsigned tile_d1,
                                                   unsigned stride0) {
  v4u g0;
  g0[0] = 1u;                                              // count=1, user mode
  g0[1] = lds_addr;                                        // LDS byte address
  g0[2] = (unsigned)(gaddr & 0xffffffffull);               // global_addr[31:0]
  g0[3] = (unsigned)((gaddr >> 32) & 0x1ffffffull) | (2u << 30);  // [56:32]|type=2
  v8i g1;
  g1[0] = (int)(1u << 16);                                 // data_size=1 (2B)
  g1[1] = (int)((tensor_d0 & 0xffffu) << 16);              // tensor_dim0[15:0]
  g1[2] = (int)((tensor_d0 >> 16) | ((tensor_d1 & 0xffffu) << 16));
  g1[3] = (int)((tensor_d1 >> 16) | (tile_d0 << 16));      // tile_dim0
  g1[4] = (int)(tile_d1 & 0xffffu);                        // tile_dim1, dim2=0
  g1[5] = (int)stride0;                                    // dim0 stride (elems)
  g1[6] = 0;
  g1[7] = 0;
  v4i z4 = {0, 0, 0, 0};
  v8i z8 = {0, 0, 0, 0, 0, 0, 0, 0};
  __builtin_amdgcn_tensor_load_to_lds(g0, g1, z4, z4, z8, 0);
}
#endif

// ---------------------------------------------------------------------------
// Tiled GEMM, bf16 in / bf16-or-f32 out. Block 256 thr = 8 waves.
// Block tile 256(M) x 64(N); each wave owns 32 rows; K chunks of 32.
// TDM double-buffered staging when available.
// ---------------------------------------------------------------------------
template <bool OUT_F32>
__global__ __launch_bounds__(256)
void gemm_bf16(const unsigned short* __restrict__ A,
               const unsigned short* __restrict__ Bm, void* __restrict__ Cout,
               int M, int N, int K) {
  __shared__ __align__(16) unsigned short As[2][256 * 32];  // 32 KB
  __shared__ __align__(16) unsigned short Bs[2][32 * 64];   //  8 KB

  const int tid = threadIdx.x;
  const int wave = tid >> 5;
  const int lane = tid & 31;
  const int ln = lane & 15;
  const int half = lane >> 4;
  const int m0 = blockIdx.y * 256;
  const int n0 = blockIdx.x * 64;

  v8f acc[2][4] = {};

#ifdef HAVE_TDM
  const unsigned long long Ab = (unsigned long long)(uintptr_t)A;
  const unsigned long long Bb = (unsigned long long)(uintptr_t)Bm;
  const unsigned asA0 = (unsigned)(uintptr_t)&As[0][0];
  const unsigned asA1 = (unsigned)(uintptr_t)&As[1][0];
  const unsigned asB0 = (unsigned)(uintptr_t)&Bs[0][0];
  const unsigned asB1 = (unsigned)(uintptr_t)&Bs[1][0];
  if (wave == 0) {  // prologue: prefetch tiles 0 and 1
    tdm_load_2d(asA0, Ab + ((size_t)m0 * K) * 2, K, M, 32, 256, K);
    tdm_load_2d(asB0, Bb + ((size_t)n0) * 2, N, K, 64, 32, N);
    if (K > 32) {
      tdm_load_2d(asA1, Ab + ((size_t)m0 * K + 32) * 2, K, M, 32, 256, K);
      tdm_load_2d(asB1, Bb + ((size_t)32 * N + n0) * 2, N, K, 64, 32, N);
    }
  }
#endif

  for (int k0 = 0; k0 < K; k0 += 32) {
    const int cur = (k0 >> 5) & 1;
#ifdef HAVE_TDM
    if (wave == 0) {
      if (k0 + 32 < K) __builtin_amdgcn_s_wait_tensorcnt(2);
      else             __builtin_amdgcn_s_wait_tensorcnt(0);
    }
    __syncthreads();
#else
    {  // fallback: cooperative synchronous staging (bf16 pairs as b32)
      const unsigned int* Ag = (const unsigned int*)A;
      const unsigned int* Bg = (const unsigned int*)Bm;
      unsigned int* AsU = (unsigned int*)&As[cur][0];
      unsigned int* BsU = (unsigned int*)&Bs[cur][0];
#pragma unroll
      for (int i = 0; i < 16; ++i) {
        int idx = tid + i * 256;              // 4096 dwords of A tile
        int r = idx >> 4, c = idx & 15;
        AsU[idx] = Ag[(((size_t)(m0 + r) * K + k0) >> 1) + c];
      }
#pragma unroll
      for (int i = 0; i < 4; ++i) {
        int idx = tid + i * 256;              // 1024 dwords of B tile
        int r = idx >> 5, c = idx & 31;
        BsU[idx] = Bg[(((size_t)(k0 + r) * N + n0) >> 1) + c];
      }
      __syncthreads();
    }
#endif

    const unsigned int* AsU = (const unsigned int*)&As[cur][0];
    const unsigned short* Bs16 = &Bs[cur][0];

    // A fragments: row = wave*32 + s*16 + ln; K pairs per documented layout
    v16bf afr[2];
#pragma unroll
    for (int s = 0; s < 2; ++s) {
      const int row = wave * 32 + s * 16 + ln;
      v8u aw;
#pragma unroll
      for (int e8 = 0; e8 < 8; ++e8) {
        int e = e8 * 2;
        int kk = e + (e & 8) + (half << 3);
        aw[e8] = AsU[row * 16 + (kk >> 1)];
      }
      afr[s] = __builtin_bit_cast(v16bf, aw);
    }
    // B fragments: col = nt*16 + ln, k = e + (e&8) + 8*half; reused by both A frags
#pragma unroll
    for (int nt = 0; nt < 4; ++nt) {
      v16us bw;
#pragma unroll
      for (int e = 0; e < 16; ++e) {
        int kk = e + (e & 8) + (half << 3);
        bw[e] = Bs16[kk * 64 + nt * 16 + ln];
      }
      v16bf bfr = __builtin_bit_cast(v16bf, bw);
      acc[0][nt] = wmma_bf16(afr[0], bfr, acc[0][nt]);
      acc[1][nt] = wmma_bf16(afr[1], bfr, acc[1][nt]);
    }
    __syncthreads();

#ifdef HAVE_TDM
    if (wave == 0 && k0 + 64 < K) {  // refill the buffer just consumed
      unsigned aA = cur ? asA1 : asA0;
      unsigned aB = cur ? asB1 : asB0;
      tdm_load_2d(aA, Ab + ((size_t)m0 * K + (k0 + 64)) * 2, K, M, 32, 256, K);
      tdm_load_2d(aB, Bb + ((size_t)(k0 + 64) * N + n0) * 2, N, K, 64, 32, N);
    }
#endif
  }

  // C/D layout: VGPR r -> row r + 8*half, col = ln
#pragma unroll
  for (int s = 0; s < 2; ++s)
#pragma unroll
    for (int nt = 0; nt < 4; ++nt)
#pragma unroll
      for (int r = 0; r < 8; ++r) {
        size_t off = (size_t)(m0 + wave * 32 + s * 16 + r + 8 * half) * N +
                     (n0 + nt * 16 + ln);
        if (OUT_F32) ((float*)Cout)[off] = acc[s][nt][r];
        else ((unsigned short*)Cout)[off] = f2bf(acc[s][nt][r]);
      }
}

// ---------------------------------------------------------------------------
// Causal flash attention, bf16 q/k/v in, bf16 y out. One wave per
// (b, h, 16-query tile); key blocks of 32; V tile staged in LDS.
// ---------------------------------------------------------------------------
__global__ __launch_bounds__(128)
void attn_fwd(const unsigned short* __restrict__ q,
              const unsigned short* __restrict__ k,
              const unsigned short* __restrict__ v,
              unsigned short* __restrict__ y) {
  __shared__ unsigned short pS[4][16 * 32];  // probabilities (bf16)  4 KB
  __shared__ unsigned short vS[4][32 * 64];  // V tile per wave      16 KB

  const int tid = threadIdx.x;
  const int wave = tid >> 5;
  const int lane = tid & 31;
  const int ln = lane & 15;
  const int half = lane >> 4;
  const int hh = blockIdx.y;
  const int bb = blockIdx.z;
  const int q0 = (blockIdx.x * 4 + wave) * 16;
  const size_t base = ((size_t)bb * T_) * HP_ + (size_t)hh * P_;
  const size_t base32 = base >> 1;
  const unsigned int* Q32 = (const unsigned int*)q;
  const unsigned int* K32 = (const unsigned int*)k;
  const unsigned int* V32 = (const unsigned int*)v;
  unsigned int* vSU = (unsigned int*)&vS[wave][0];
  const unsigned int* pSU = (const unsigned int*)&pS[wave][0];

  // Q fragments (two K-chunks over P=64), unscaled (1/sqrt(P) folded into S)
  v16bf qa[2];
#pragma unroll
  for (int c = 0; c < 2; ++c) {
    v8u w;
#pragma unroll
    for (int e8 = 0; e8 < 8; ++e8) {
      int e = e8 * 2;
      int pp = c * 32 + e + (e & 8) + (half << 3);
      w[e8] = Q32[base32 + (size_t)(q0 + ln) * (HP_ / 2) + (pp >> 1)];
    }
    qa[c] = __builtin_bit_cast(v16bf, w);
  }

  v8f o[4] = {};
  float mi[8], li[8];
#pragma unroll
  for (int r = 0; r < 8; ++r) { mi[r] = -3.0e38f; li[r] = 0.0f; }

  for (int kb = 0; kb < q0 + 16; kb += 32) {
    // Stage V tile [32 keys x 64 dims] (coalesced b32 loads)
#pragma unroll
    for (int i = 0; i < 32; ++i)
      vSU[i * 32 + lane] = V32[base32 + (size_t)(kb + i) * (HP_ / 2) + lane];

    // S = Q * K^T for two 16-key tiles
    v8f s0 = {}, s1 = {};
#pragma unroll
    for (int c = 0; c < 2; ++c) {
      v8u w0, w1;
#pragma unroll
      for (int e8 = 0; e8 < 8; ++e8) {
        int e = e8 * 2;
        int pp = c * 32 + e + (e & 8) + (half << 3);
        w0[e8] = K32[base32 + (size_t)(kb + ln) * (HP_ / 2) + (pp >> 1)];
        w1[e8] = K32[base32 + (size_t)(kb + 16 + ln) * (HP_ / 2) + (pp >> 1)];
      }
      s0 = wmma_bf16(qa[c], __builtin_bit_cast(v16bf, w0), s0);
      s1 = wmma_bf16(qa[c], __builtin_bit_cast(v16bf, w1), s1);
    }

    // Causal mask, scale, online softmax (row = q0 + r + 8*half)
#pragma unroll
    for (int r = 0; r < 8; ++r) {
      int qrow = q0 + r + 8 * half;
      float v0 = (kb + ln      <= qrow) ? s0[r] * 0.125f : -3.0e38f;
      float v1 = (kb + 16 + ln <= qrow) ? s1[r] * 0.125f : -3.0e38f;
      float mx = fmaxf(v0, v1);
      mx = fmaxf(mx, __shfl_xor(mx, 1, 32));
      mx = fmaxf(mx, __shfl_xor(mx, 2, 32));
      mx = fmaxf(mx, __shfl_xor(mx, 4, 32));
      mx = fmaxf(mx, __shfl_xor(mx, 8, 32));
      float mnew = fmaxf(mi[r], mx);
      float corr = __expf(mi[r] - mnew);
      float p0 = __expf(v0 - mnew);
      float p1 = __expf(v1 - mnew);
      float rs = p0 + p1;
      rs += __shfl_xor(rs, 1, 32);
      rs += __shfl_xor(rs, 2, 32);
      rs += __shfl_xor(rs, 4, 32);
      rs += __shfl_xor(rs, 8, 32);
      li[r] = li[r] * corr + rs;
      mi[r] = mnew;
#pragma unroll
      for (int nt = 0; nt < 4; ++nt) o[nt][r] *= corr;
      pS[wave][(r + 8 * half) * 32 + ln]      = f2bf(p0);
      pS[wave][(r + 8 * half) * 32 + 16 + ln] = f2bf(p1);
    }
    // cross-lane RAW through LDS within this wave (CDNA5 split counter)
    asm volatile("s_wait_dscnt 0" ::: "memory");

    // P fragment (A layout): b32 pair reads from LDS
    v8u pw;
#pragma unroll
    for (int e8 = 0; e8 < 8; ++e8) {
      int e = e8 * 2;
      int kk = e + (e & 8) + (half << 3);
      pw[e8] = pSU[ln * 16 + (kk >> 1)];
    }
    v16bf pa = __builtin_bit_cast(v16bf, pw);

    // O += P * V   (B fragment: key k = e + (e&8) + 8*half, col = nt*16 + ln)
#pragma unroll
    for (int nt = 0; nt < 4; ++nt) {
      v16us vw;
#pragma unroll
      for (int e = 0; e < 16; ++e) {
        int kk = e + (e & 8) + (half << 3);
        vw[e] = vS[wave][kk * 64 + nt * 16 + ln];
      }
      o[nt] = wmma_bf16(pa, __builtin_bit_cast(v16bf, vw), o[nt]);
    }
  }

  // Normalize and store y (bf16)
#pragma unroll
  for (int r = 0; r < 8; ++r) {
    float inv = 1.0f / li[r];
#pragma unroll
    for (int nt = 0; nt < 4; ++nt)
      y[base + (size_t)(q0 + r + 8 * half) * HP_ + (nt * 16 + ln)] =
          f2bf(o[nt][r] * inv);
  }
}

// ---------------------------------------------------------------------------
extern "C" void kernel_launch(void* const* d_in, const int* in_sizes, int n_in,
                              void* d_out, int out_size, void* d_ws,
                              size_t ws_size, hipStream_t stream) {
  const float* x  = (const float*)d_in[0];
  // d_in[1] = attn_mask (causal; applied analytically in attn_fwd)
  const float* Wq = (const float*)d_in[2];
  const float* Wk = (const float*)d_in[3];
  const float* Wv = (const float*)d_in[4];
  const float* Wo = (const float*)d_in[5];
  float* out = (float*)d_out;

  const int M = B_ * T_;              // 8192
  const int N = HP_;                  // 1024
  const int K = D_;                   // 1024
  const size_t nx = (size_t)M * K;    // 8 M elements
  const size_t nw = (size_t)K * N;    // 1 M elements

  unsigned short* xb  = (unsigned short*)d_ws;
  unsigned short* Wqb = xb + nx;
  unsigned short* Wkb = Wqb + nw;
  unsigned short* Wvb = Wkb + nw;
  unsigned short* Wob = Wvb + nw;
  unsigned short* qb  = Wob + nw;
  unsigned short* kb  = qb + nx;
  unsigned short* vb  = kb + nx;
  unsigned short* yb  = vb + nx;      // total ~88 MB of d_ws

  hipLaunchKernelGGL(cvt_f32_bf16, dim3((nx + 1023) / 1024), dim3(256), 0,
                     stream, x, xb, (int)nx);
  hipLaunchKernelGGL(cvt_f32_bf16, dim3((nw + 1023) / 1024), dim3(256), 0,
                     stream, Wq, Wqb, (int)nw);
  hipLaunchKernelGGL(cvt_f32_bf16, dim3((nw + 1023) / 1024), dim3(256), 0,
                     stream, Wk, Wkb, (int)nw);
  hipLaunchKernelGGL(cvt_f32_bf16, dim3((nw + 1023) / 1024), dim3(256), 0,
                     stream, Wv, Wvb, (int)nw);
  hipLaunchKernelGGL(cvt_f32_bf16, dim3((nw + 1023) / 1024), dim3(256), 0,
                     stream, Wo, Wob, (int)nw);

  dim3 blk(256);
  dim3 grd(N / 64, M / 256);
  hipLaunchKernelGGL((gemm_bf16<false>), grd, blk, 0, stream, xb, Wqb, qb, M, N, K);
  hipLaunchKernelGGL((gemm_bf16<false>), grd, blk, 0, stream, xb, Wkb, kb, M, N, K);
  hipLaunchKernelGGL((gemm_bf16<false>), grd, blk, 0, stream, xb, Wvb, vb, M, N, K);
  hipLaunchKernelGGL(attn_fwd, dim3(T_ / 64, H_, B_), dim3(128), 0, stream,
                     qb, kb, vb, yb);
  hipLaunchKernelGGL((gemm_bf16<true>), dim3(D_ / 64, M / 256), blk, 0, stream,
                     yb, Wob, out, M, D_, N);
}
